// MolSpaceMultiHeadAttention_87978110091597
// MI455X (gfx1250) — matrix operand order repaired and checked by reading, not compiled, not run on hardware
//
#include <hip/hip_runtime.h>
#include <math.h>

// ---- problem constants (match reference) ----
#define T_TOK  131072
#define EMB    1024
#define NHEAD  8
#define DKH    128              // per-head dim (1024/8)
#define TM     64               // tokens per block (4 WMMA M-tiles)
#define NWAVE  16               // 512 threads
#define XS     (EMB + 8)        // LDS row stride in bf16 elems (pad vs bank conflicts)

// packed-weight buffer: [3 mats][64 ntiles][32 kchunks][32 lanes][16 bf16]
#define NTILES_N    64
#define KCHUNKS     32
#define FRAGS_TOTAL (3 * NTILES_N * KCHUNKS * 32)   // 196,608 fragments of 16 bf16
#define PACK_BYTES  ((size_t)FRAGS_TOTAL * 16 * 2)  // 6,291,456 B

typedef __bf16 bf16;
typedef bf16  v16bf __attribute__((ext_vector_type(16)));
typedef float v8f   __attribute__((ext_vector_type(8)));

// A-fragment (16x32 bf16) in this code's contiguous-k order (B matches):
// lane = hf*16 + m ; elems 0..15 -> K = hf*16 + 0..15
__device__ __forceinline__ v16bf load_a_frag(const bf16* __restrict__ p) {
    v16bf a;
#pragma unroll
    for (int e = 0; e < 8; ++e) a[e] = p[e];
#pragma unroll
    for (int e = 0; e < 8; ++e) a[8 + e] = p[8 + e];
    return a;
}

// B-fragment (32x16 bf16), f32 fallback path:
__device__ __forceinline__ v16bf load_b_frag_f32(const float* __restrict__ wp) {
    v16bf b;
#pragma unroll
    for (int e = 0; e < 16; ++e) b[e] = (bf16)wp[e];
    return b;
}

// One 16-col output tile over 4 M-tiles (64 tokens):
// dst[row*XS + l16] (dst already offset by column base), dst gets bf16 results.
template<bool PACKED>
__device__ __forceinline__ void gemm_tile4(const bf16* __restrict__ shx,
                                           const float* __restrict__ Wf,   // f32 path
                                           const v16bf* __restrict__ Wp,   // packed path
                                           bf16* __restrict__ dst,
                                           int ntile, int lane) {
    const int hf  = lane >> 4;
    const int l16 = lane & 15;

    v8f acc[4];
#pragma unroll
    for (int mt = 0; mt < 4; ++mt) acc[mt] = (v8f){0.f,0.f,0.f,0.f,0.f,0.f,0.f,0.f};

    const float* wrow = PACKED ? nullptr
                               : (Wf + (ntile * 16 + l16) * EMB + hf * 16);
    const bf16* ab = shx + l16 * XS + hf * 16;

#pragma unroll 2
    for (int kc = 0; kc < KCHUNKS; ++kc) {
        v16bf b;
        if constexpr (PACKED)
            b = Wp[kc * 32 + lane];               // lane*32B contiguous: coalesced b128 x2
        else
            b = load_b_frag_f32(wrow + kc * 32);
#pragma unroll
        for (int mt = 0; mt < 4; ++mt) {
            v16bf a = load_a_frag(ab + mt * 16 * XS + kc * 32);
            acc[mt] = __builtin_amdgcn_wmma_f32_16x16x32_bf16(false, a, false, b,
                                                              (short)0, acc[mt], false, false);
        }
    }
    // D layout: reg r -> row (mt*16 + r + 8*hf), col = l16 (caller pre-offset col base)
#pragma unroll
    for (int mt = 0; mt < 4; ++mt)
#pragma unroll
        for (int r = 0; r < 8; ++r)
            dst[(mt * 16 + r + 8 * hf) * XS + l16] = (bf16)acc[mt][r];
}

// One-time weight repack: f32 [N][K] row-major -> bf16 WMMA-fragment order
// wpk[(mat*64+ntile)*1024 + kc*32 + lane][e] = W[ntile*16+(lane&15)][kc*32+(lane>>4)*16+e]
__global__ __launch_bounds__(256)
void repack_weights_kernel(const float* __restrict__ Wq,
                           const float* __restrict__ Wk,
                           const float* __restrict__ Wv,
                           v16bf* __restrict__ wpk) {
    const int g     = blockIdx.x * 256 + threadIdx.x;
    const int lane  = g & 31;
    const int kc    = (g >> 5) & 31;
    const int ntile = (g >> 10) & 63;
    const int mat   = g >> 16;                           // uniform per block
    const float* W  = (mat == 0) ? Wq : (mat == 1) ? Wk : Wv;
    const float* src = W + (ntile * 16 + (lane & 15)) * EMB + kc * 32 + (lane >> 4) * 16;
    v16bf b;
#pragma unroll
    for (int e = 0; e < 16; ++e) b[e] = (bf16)src[e];
    wpk[g] = b;
}

template<bool PACKED>
__global__ __launch_bounds__(512, 1)
void MolSpaceMHA_fused_kernel(const float* __restrict__ x,
                              const float* __restrict__ Wq,
                              const float* __restrict__ Wk,
                              const float* __restrict__ Wv,
                              const v16bf* __restrict__ wpk,
                              float* __restrict__ out) {
    // 274 KB of LDS (CDNA5: 320 KB/WGP). sh_qk holds {q-half | k-half} during
    // score passes, then is reused as the full v buffer.
    __shared__ bf16  sh_x[TM * XS];        // 132,096 B
    __shared__ bf16  sh_qk[TM * XS];       // 132,096 B (cols 0..511 q-half, 512..1023 k-half)
    __shared__ float sh_attn[TM * NHEAD * NHEAD];   // 16,384 B

    const int  tid  = threadIdx.x;          // 0..511
    const int  wave = tid >> 5;             // 0..15
    const int  lane = tid & 31;
    const long t0   = (long)blockIdx.x * TM;

    // ---- load 64x1024 x-tile, convert f32 -> bf16 into LDS ----
    {
        const float4* xv = (const float4*)(x + t0 * EMB);
        for (int i = tid; i < TM * EMB / 4; i += 512) {
            int row  = i >> 8;               // 256 float4 per row
            int col4 = i & 255;
            float4 f = xv[row * (EMB / 4) + col4];
            bf16* d  = sh_x + row * XS + col4 * 4;
            d[0] = (bf16)f.x; d[1] = (bf16)f.y; d[2] = (bf16)f.z; d[3] = (bf16)f.w;
        }
    }
    __syncthreads();

    // per-thread score accumulators: thread owns (t = tid>>3, h = tid&7)
    const int st = tid >> 3;
    const int sh = tid & 7;
    float s[NHEAD];
#pragma unroll
    for (int g = 0; g < NHEAD; ++g) s[g] = 0.f;

    // ---- stages 1+2: q per 4-head half, k per 4-head half, fused scores ----
#pragma unroll
    for (int hg = 0; hg < 2; ++hg) {
        // q-half GEMM: 32 ntiles (heads hg*4..hg*4+3) -> sh_qk cols [0,512)
#pragma unroll
        for (int i = 0; i < 2; ++i) {
            const int j     = wave + 16 * i;          // 0..31
            const int ntile = hg * 32 + j;
            gemm_tile4<PACKED>(sh_x, Wq,
                               PACKED ? (wpk + (size_t)ntile * (KCHUNKS * 32)) : nullptr,
                               sh_qk + j * 16, ntile, lane);
        }
        __syncthreads();

#pragma unroll
        for (int gg = 0; gg < 2; ++gg) {
            // k-half GEMM: 32 ntiles (heads gg*4..gg*4+3) -> sh_qk cols [512,1024)
#pragma unroll
            for (int i = 0; i < 2; ++i) {
                const int j     = wave + 16 * i;      // 0..31
                const int ntile = gg * 32 + j;
                gemm_tile4<PACKED>(sh_x, Wk,
                                   PACKED ? (wpk + (size_t)(64 + ntile) * (KCHUNKS * 32)) : nullptr,
                                   sh_qk + 512 + j * 16, ntile, lane);
            }
            __syncthreads();

            // score pass: threads whose head is in this q-half accumulate 4 g's
            if ((sh >> 2) == hg) {
                const bf16* qp = sh_qk + st * XS + (sh & 3) * DKH;
                const bf16* kp = sh_qk + st * XS + 512;
#pragma unroll
                for (int gl = 0; gl < 4; ++gl) {
                    float acc = 0.f;
#pragma unroll 16
                    for (int d = 0; d < DKH; ++d)
                        acc += (float)qp[d] * (float)kp[gl * DKH + d];
                    s[gg * 4 + gl] = acc * 0.088388347648318447f;   // 1/sqrt(128)
                }
            }
            __syncthreads();    // before k-half (or q-half) is overwritten
        }
    }

    // ---- softmax in registers, publish attn ----
    {
        float mx = s[0];
#pragma unroll
        for (int g = 1; g < NHEAD; ++g) mx = fmaxf(mx, s[g]);
        float sum = 0.f;
#pragma unroll
        for (int g = 0; g < NHEAD; ++g) { s[g] = __expf(s[g] - mx); sum += s[g]; }
        const float inv = 1.f / sum;
        float* ap = sh_attn + (st * NHEAD + sh) * NHEAD;
#pragma unroll
        for (int g = 0; g < NHEAD; ++g) ap[g] = s[g] * inv;
    }
    // no barrier needed before v-GEMM: sh_attn is disjoint from sh_qk, and the
    // last score pass was already followed by a barrier.

    // ---- stage 3: v = x Wv^T (64 ntiles over 16 waves) into sh_qk (full) ----
#pragma unroll
    for (int i = 0; i < 4; ++i) {
        const int j = wave + 16 * i;                  // 0..63
        gemm_tile4<PACKED>(sh_x, Wv,
                           PACKED ? (wpk + (size_t)(128 + j) * (KCHUNKS * 32)) : nullptr,
                           sh_qk + j * 16, j, lane);
    }
    __syncthreads();

    // ---- stage 4: out[t, h*128+d] = sum_g attn[t,h,g] * v[t, g*128+d] ----
    {
        float4* ov = (float4*)(out + t0 * EMB);
        for (int i = tid; i < TM * EMB / 4; i += 512) {
            const int t  = i >> 8;
            const int c4 = i & 255;
            const int n  = c4 * 4;
            const int h  = n >> 7;
            const int d  = n & 127;
            const float* ap = sh_attn + (t * NHEAD + h) * NHEAD;
            const bf16*  vp = sh_qk + t * XS + d;
            float4 o = {0.f, 0.f, 0.f, 0.f};
#pragma unroll
            for (int g = 0; g < NHEAD; ++g) {
                const float a  = ap[g];
                const bf16* vv = vp + g * DKH;
                o.x += a * (float)vv[0];
                o.y += a * (float)vv[1];
                o.z += a * (float)vv[2];
                o.w += a * (float)vv[3];
            }
            ov[t * (EMB / 4) + c4] = o;
        }
    }
}

extern "C" void kernel_launch(void* const* d_in, const int* in_sizes, int n_in,
                              void* d_out, int out_size, void* d_ws, size_t ws_size,
                              hipStream_t stream) {
    (void)in_sizes; (void)n_in; (void)out_size;
    const float* x  = (const float*)d_in[0];
    const float* Wq = (const float*)d_in[1];
    const float* Wk = (const float*)d_in[2];
    const float* Wv = (const float*)d_in[3];
    float* out = (float*)d_out;

    const dim3 grid(T_TOK / TM);   // 2048 blocks, 64 tokens each
    const dim3 block(512);         // 16 wave32 waves

    if (d_ws != nullptr && ws_size >= PACK_BYTES) {
        v16bf* wpk = (v16bf*)d_ws;
        repack_weights_kernel<<<dim3(FRAGS_TOTAL / 256), dim3(256), 0, stream>>>(Wq, Wk, Wv, wpk);
        MolSpaceMHA_fused_kernel<true><<<grid, block, 0, stream>>>(x, Wq, Wk, Wv, wpk, out);
    } else {
        MolSpaceMHA_fused_kernel<false><<<grid, block, 0, stream>>>(x, Wq, Wk, Wv, nullptr, out);
    }
}
